// PMATItemEncoder_9423158247487
// MI455X (gfx1250) — matrix operand order repaired and checked by reading, not compiled, not run on hardware
//
#include <hip/hip_runtime.h>

// ---------------------------------------------------------------------------
// Problem constants
// ---------------------------------------------------------------------------
#define NROWS   65536        // B*S = 64*1024
#define TXT_K   768
#define VIS_K   512
#define HID     256
#define CB      1024
#define L_ID    4
#define FUSE_K  512          // 2*HID

typedef __attribute__((ext_vector_type(16))) __bf16 v16bf;
typedef __attribute__((ext_vector_type(8)))  float  v8f;

union Frag {
    v16bf    v;
    unsigned u[8];
    uint4    q[2];
};

__device__ __forceinline__ v8f zero8() {
    v8f z;
#pragma unroll
    for (int i = 0; i < 8; ++i) z[i] = 0.0f;
    return z;
}

__device__ __forceinline__ unsigned short f2bf(float x) {
    union { float f; unsigned u; } v; v.f = x;
    unsigned r = v.u + 0x7FFFu + ((v.u >> 16) & 1u);   // round-to-nearest-even
    return (unsigned short)(r >> 16);
}
__device__ __forceinline__ unsigned pack2(float a, float b) {
    return (unsigned)f2bf(a) | ((unsigned)f2bf(b) << 16);
}

// Stage a [rows x 32] f32 tile into LDS as bf16, row stride 40 shorts (80B pad
// keeps 16B alignment for ds_load_b128 fragments and staggers banks).
__device__ __forceinline__ void stage_a_f32(const float* __restrict__ A, int ldA,
                                            int row0, int k0,
                                            unsigned short* sA, int rows,
                                            int tid, int nthreads) {
    int total = rows * 32;
    for (int i = tid * 4; i < total; i += nthreads * 4) {
        int r = i >> 5, c = i & 31;
        const float4 f = *(const float4*)(A + (size_t)(row0 + r) * ldA + k0 + c);
        uint2 p; p.x = pack2(f.x, f.y); p.y = pack2(f.z, f.w);
        *(uint2*)(sA + r * 40 + c) = p;
    }
}

// Async global->LDS copy of a [rows x 32] bf16 tile (16B per active lane).
// GLOBAL_LOAD_ASYNC_TO_LDS_B128: VDST = per-lane LDS byte address, VADDR = 64b
// global address; tracked by ASYNCcnt (must s_wait_asynccnt before readers).
__device__ __forceinline__ void async_stage_bf16(const unsigned short* __restrict__ A,
                                                 int ldA, int row0, int k0,
                                                 unsigned short* sA, int rows,
                                                 int tid, int nthreads) {
    int total = rows * 32;
    for (int i = tid * 8; i < total; i += nthreads * 8) {
        int r = i >> 5, c = i & 31;
        const unsigned short* g = A + (size_t)(row0 + r) * ldA + k0 + c;
        unsigned lds = (unsigned)(unsigned long long)(sA + r * 40 + c);
        asm volatile("global_load_async_to_lds_b128 %0, %1, off"
                     :: "v"(lds), "v"(g) : "memory");
    }
}
__device__ __forceinline__ void wait_async0() {
    asm volatile("s_wait_asynccnt 0x0" ::: "memory");
}

// A-fragment (16x32 bf16) per ISA 7.12.2: lane<16 -> V0..3 K=0..7, V4..7 K=16..23
//                                          lane>=16 -> V0..3 K=8..15, V4..7 K=24..31
__device__ __forceinline__ void load_a_frag(const unsigned short* sA, int rowBase,
                                            int lane, Frag& f) {
    int m  = lane & 15;
    int kb = (lane >> 4) * 8;
    const unsigned short* p = sA + (rowBase + m) * 40 + kb;
    f.q[0] = *(const uint4*)p;          // K = kb .. kb+7
    f.q[1] = *(const uint4*)(p + 16);   // K = kb+16 .. kb+23
}

// B-fragment from pre-packed global layout: frag = kt*NT + nt, 256 dwords/frag.
__device__ __forceinline__ void load_b_frag(const unsigned* __restrict__ Bp,
                                            int fragIdx, int lane, Frag& f) {
    const unsigned* p = Bp + (size_t)fragIdx * 256 + lane * 8;
    f.q[0] = *(const uint4*)p;
    f.q[1] = *(const uint4*)(p + 4);
}

#define WMMA_BF16(A, B, C) \
    __builtin_amdgcn_wmma_f32_16x16x32_bf16(false, (A), false, (B), (short)0, (C), false, false)

// ---------------------------------------------------------------------------
// Kernel: pack a K x Nout f32 matrix into WMMA B-fragment order (bf16).
// element(k,n) = src[k*ldK + n*ldN].  One 256-thread block per fragment.
// ---------------------------------------------------------------------------
__global__ __launch_bounds__(256) void k_pack(const float* __restrict__ src,
                                              int ldK, int ldN, int NT,
                                              unsigned* __restrict__ dst) {
    int frag = blockIdx.x;
    int kt = frag / NT, nt = frag % NT;
    int tid  = threadIdx.x;
    int lane = tid >> 3;          // 0..31
    int v    = tid & 7;           // 0..7
    int n = nt * 16 + (lane & 15);
    int k = kt * 32 + (lane >> 4) * 8 + (v >> 2) * 8 + v * 2;
    float a = src[(size_t)k * ldK + (size_t)n * ldN];
    float b = src[(size_t)(k + 1) * ldK + (size_t)n * ldN];
    dst[(size_t)frag * 256 + lane * 8 + v] = pack2(a, b);
}

// ---------------------------------------------------------------------------
// Kernel: codebook squared norms  cnorm[l*CB+c] = sum_k C[l,c,k]^2
// ---------------------------------------------------------------------------
__global__ __launch_bounds__(256) void k_cnorm(const float* __restrict__ cb,
                                               float* __restrict__ cnorm) {
    int i = blockIdx.x * 256 + threadIdx.x;           // 0..4095
    const float* p = cb + (size_t)i * HID;
    float s = 0.f;
    for (int k = 0; k < HID; k += 4) {
        float4 f = *(const float4*)(p + k);
        s += f.x * f.x + f.y * f.y + f.z * f.z + f.w * f.w;
    }
    cnorm[i] = s;
}

__global__ __launch_bounds__(256) void k_zero(int* __restrict__ hist,
                                              float* __restrict__ accum) {
    int i = blockIdx.x * 256 + threadIdx.x;
    if (i < L_ID * CB) hist[i] = 0;
    if (i < 16)        accum[i] = 0.f;
}

// ---------------------------------------------------------------------------
// Kernel: multimodal encoder + softmax-weighted fusion + L2 normalize.
// 32 rows / workgroup; 8 waves = 2 row-groups x 4 col-groups (64 cols each).
// Double-buffered LDS A staging: one barrier per k-tile.
// ---------------------------------------------------------------------------
__global__ __launch_bounds__(256) void k_encoder(
        const float* __restrict__ text, const float* __restrict__ vis,
        const unsigned* __restrict__ WtP, const unsigned* __restrict__ WvP,
        const float* __restrict__ b_text, const float* __restrict__ b_vis,
        const float* __restrict__ modal_w,
        float* __restrict__ fused, float* __restrict__ residual,
        float* __restrict__ quantized) {
    __shared__ __align__(16) unsigned short sA[2][32 * 40];
    __shared__ float sOut[32 * 256];
    __shared__ float sRed[32 * 8];

    int tid = threadIdx.x, lane = tid & 31, w = tid >> 5;
    int rg = w >> 2, cg = w & 3;
    int row0 = blockIdx.x * 32;

    float m0 = modal_w[0], m1 = modal_w[1];
    float mx = fmaxf(m0, m1);
    float e0 = __expf(m0 - mx), e1 = __expf(m1 - mx);
    float w0 = e0 / (e0 + e1), w1 = e1 / (e0 + e1);

    for (int mod = 0; mod < 2; ++mod) {
        const float* A = mod ? vis : text;
        const unsigned* Bp = mod ? WvP : WtP;
        int K  = mod ? VIS_K : TXT_K;
        int KT = K >> 5;
        v8f acc[4];
#pragma unroll
        for (int j = 0; j < 4; ++j) acc[j] = zero8();

        __syncthreads();                               // protect sA[0] reuse
        stage_a_f32(A, K, row0, 0, sA[0], 32, tid, 256);
        __syncthreads();

        for (int kt = 0; kt < KT; ++kt) {
            const unsigned short* cur = sA[kt & 1];
            if (kt + 1 < KT) {
                stage_a_f32(A, K, row0, (kt + 1) * 32, sA[(kt + 1) & 1], 32, tid, 256);
                __builtin_prefetch(A + (size_t)(row0 + (tid & 31)) * K + (kt + 1) * 32, 0, 1);
            }
            Frag a; load_a_frag(cur, rg * 16, lane, a);
            Frag b[4];
#pragma unroll
            for (int j = 0; j < 4; ++j)
                load_b_frag(Bp, kt * 16 + (cg * 4 + j), lane, b[j]);
#pragma unroll
            for (int j = 0; j < 4; ++j)
                acc[j] = WMMA_BF16(a.v, b[j].v, acc[j]);
            __syncthreads();
        }
        float wgt = mod ? w1 : w0;
        int rbase = rg * 16 + ((lane >> 4) ? 8 : 0);
#pragma unroll
        for (int j = 0; j < 4; ++j) {
            int n = (cg * 4 + j) * 16 + (lane & 15);
#pragma unroll
            for (int v = 0; v < 8; ++v) {
                float val = wgt * acc[j][v];
                if (mod == 0) sOut[(rbase + v) * 256 + n]  = val;
                else          sOut[(rbase + v) * 256 + n] += val;
            }
        }
    }
    __syncthreads();

    // bias + row L2-normalize, write fused / residual / quantized(=0)
    int r = tid >> 3, seg = tid & 7;
    float ss = 0.f;
    for (int c = seg * 32; c < seg * 32 + 32; ++c) {
        float val = sOut[r * 256 + c] + w0 * b_text[c] + w1 * b_vis[c];
        sOut[r * 256 + c] = val;
        ss += val * val;
    }
    sRed[r * 8 + seg] = ss;
    __syncthreads();
    float tot = 0.f;
#pragma unroll
    for (int i = 0; i < 8; ++i) tot += sRed[r * 8 + i];
    float scale = 1.0f / fmaxf(sqrtf(tot), 1e-12f);
    size_t gb = (size_t)(row0 + r) * 256;
    for (int c = seg * 32; c < seg * 32 + 32; c += 4) {
        float4 v4;
        v4.x = sOut[r * 256 + c + 0] * scale;
        v4.y = sOut[r * 256 + c + 1] * scale;
        v4.z = sOut[r * 256 + c + 2] * scale;
        v4.w = sOut[r * 256 + c + 3] * scale;
        *(float4*)(fused    + gb + c) = v4;
        *(float4*)(residual + gb + c) = v4;
        float4 z; z.x = z.y = z.z = z.w = 0.f;
        *(float4*)(quantized + gb + c) = z;
    }
}

// ---------------------------------------------------------------------------
// Kernel: one RVQ layer.  32 rows / workgroup; full 32x256 residual tile
// staged once to LDS (bf16), 1024 codebook columns swept in two halves
// (8 waves = 2 row-groups x 4 col-groups x 8 n-tiles per half).
// Halves per-row codebook L2 traffic vs 16-row blocks.
// ---------------------------------------------------------------------------
__global__ __launch_bounds__(256) void k_rvq(
        const unsigned* __restrict__ cbP,   // packed codebook frags, this layer
        const float* __restrict__ cb,       // raw codebook [CB][HID], this layer
        const float* __restrict__ cnorm,    // [CB], this layer
        float* __restrict__ residual, float* __restrict__ quantized,
        int* __restrict__ ids_out, int layer,
        float* __restrict__ loss_accum, int* __restrict__ hist) {
    __shared__ __align__(16) unsigned short sA[8][32 * 40];   // full 32x256 tile
    __shared__ float sBV[8][16];
    __shared__ int   sBI[8][16];
    __shared__ int   sIdx[32];
    __shared__ float sLoss[256];

    int tid = threadIdx.x, lane = tid & 31, w = tid >> 5;
    int rg = w >> 2, cg = w & 3;
    int row0 = blockIdx.x * 32;

    // stage entire residual tile as bf16 (8 k-tiles), one barrier
#pragma unroll
    for (int kt = 0; kt < 8; ++kt)
        stage_a_f32(residual, HID, row0, kt * 32, sA[kt], 32, tid, 256);
    __syncthreads();

    float bv[8]; int bi[8];
#pragma unroll
    for (int v = 0; v < 8; ++v) { bv[v] = -3.4e38f; bi[v] = 0; }

    for (int h = 0; h < 2; ++h) {
        v8f acc[8];
#pragma unroll
        for (int j = 0; j < 8; ++j) acc[j] = zero8();
#pragma unroll
        for (int kt = 0; kt < 8; ++kt) {
            Frag a; load_a_frag(sA[kt], rg * 16, lane, a);
            Frag b[8];
#pragma unroll
            for (int j = 0; j < 8; ++j)
                load_b_frag(cbP, kt * 64 + h * 32 + cg * 8 + j, lane, b[j]);
#pragma unroll
            for (int j = 0; j < 8; ++j)
                acc[j] = WMMA_BF16(a.v, b[j].v, acc[j]);
        }
#pragma unroll
        for (int j = 0; j < 8; ++j) {
            int n = (h * 32 + cg * 8 + j) * 16 + (lane & 15);
            float cn = 0.5f * cnorm[n];
#pragma unroll
            for (int v = 0; v < 8; ++v) {
                float s = acc[j][v] - cn;
                if (s > bv[v] || (s == bv[v] && n < bi[v])) { bv[v] = s; bi[v] = n; }
            }
        }
    }
    // butterfly max across the 16 lanes sharing each row half
#pragma unroll
    for (int off = 1; off < 16; off <<= 1) {
#pragma unroll
        for (int v = 0; v < 8; ++v) {
            float ov = __shfl_xor(bv[v], off, 32);
            int   oi = __shfl_xor(bi[v], off, 32);
            if (ov > bv[v] || (ov == bv[v] && oi < bi[v])) { bv[v] = ov; bi[v] = oi; }
        }
    }
    if ((lane & 15) == 0) {
        int rb = (lane >> 4) * 8;
#pragma unroll
        for (int v = 0; v < 8; ++v) { sBV[w][rb + v] = bv[v]; sBI[w][rb + v] = bi[v]; }
    }
    __syncthreads();
    if (tid < 32) {
        int rgi = tid >> 4, r16 = tid & 15;
        float best = -3.4e38f; int bidx = 0;
        for (int c4 = 0; c4 < 4; ++c4) {
            int ww = rgi * 4 + c4;
            float v = sBV[ww][r16];
            int   i = sBI[ww][r16];
            if (v > best || (v == best && i < bidx)) { best = v; bidx = i; }
        }
        sIdx[tid] = bidx;
        ids_out[(size_t)(row0 + tid) * L_ID + layer] = bidx;
        atomicAdd(&hist[bidx], 1);
    }
    __syncthreads();

    // update residual / quantized, accumulate commitment loss partial
    int r = tid >> 3, seg = tid & 7;
    int idx = sIdx[r];
    float lsum = 0.f;
    size_t gb = (size_t)(row0 + r) * HID + seg * 32;
    const float* cq = cb + (size_t)idx * HID + seg * 32;
    for (int c = 0; c < 32; c += 4) {
        float4 rr = *(const float4*)(residual + gb + c);
        float4 qq = *(const float4*)(cq + c);
        float4 nr; nr.x = rr.x - qq.x; nr.y = rr.y - qq.y;
        nr.z = rr.z - qq.z; nr.w = rr.w - qq.w;
        *(float4*)(residual + gb + c) = nr;
        float4 qa = *(const float4*)(quantized + gb + c);
        qa.x += qq.x; qa.y += qq.y; qa.z += qq.z; qa.w += qq.w;
        *(float4*)(quantized + gb + c) = qa;
        lsum += nr.x * nr.x + nr.y * nr.y + nr.z * nr.z + nr.w * nr.w;
    }
    sLoss[tid] = lsum;
    __syncthreads();
    for (int s = 128; s > 0; s >>= 1) {
        if (tid < s) sLoss[tid] += sLoss[tid + s];
        __syncthreads();
    }
    if (tid == 0) atomicAdd(loss_accum, sLoss[0]);
}

// ---------------------------------------------------------------------------
// Kernel: q = LN(0.7*quant + 0.3*fused); recon loss; emit combined bf16 [N,512]
// ---------------------------------------------------------------------------
__global__ __launch_bounds__(256) void k_mix(
        const float* __restrict__ fused, float* __restrict__ quantized,
        unsigned short* __restrict__ combined, float* __restrict__ recon_accum) {
    __shared__ float sSum[32 * 8], sSq[32 * 8], sLoss[256];
    int tid = threadIdx.x;
    int r = tid >> 3, seg = tid & 7;
    int row0 = blockIdx.x * 32;
    size_t gb = (size_t)(row0 + r) * 256 + seg * 32;
    float s = 0.f, s2 = 0.f, rec = 0.f;
    for (int c = 0; c < 32; ++c) {
        float f  = fused[gb + c];
        float qn = quantized[gb + c];
        float d  = qn - f;
        rec += d * d;
        float q = 0.7f * qn + 0.3f * f;
        s += q; s2 += q * q;
    }
    sSum[r * 8 + seg] = s; sSq[r * 8 + seg] = s2; sLoss[tid] = rec;
    __syncthreads();
    float ts = 0.f, ts2 = 0.f;
#pragma unroll
    for (int i = 0; i < 8; ++i) { ts += sSum[r * 8 + i]; ts2 += sSq[r * 8 + i]; }
    float mu  = ts * (1.0f / 256.0f);
    float var = ts2 * (1.0f / 256.0f) - mu * mu;
    float inv = rsqrtf(var + 1e-5f);
    size_t cbase = (size_t)(row0 + r) * 512;
    for (int c = 0; c < 32; ++c) {
        float f  = fused[gb + c];
        float qn = quantized[gb + c];
        float q  = 0.7f * qn + 0.3f * f;
        float y  = (q - mu) * inv;
        quantized[gb + c] = y;                              // quantized_out
        combined[cbase + seg * 32 + c]       = f2bf(f);
        combined[cbase + 256 + seg * 32 + c] = f2bf(y);
    }
    __syncthreads();
    for (int st = 128; st > 0; st >>= 1) {
        if (tid < st) sLoss[tid] += sLoss[tid + st];
        __syncthreads();
    }
    if (tid == 0) atomicAdd(recon_accum, sLoss[0]);
}

// ---------------------------------------------------------------------------
// Kernel: item = gelu(LN(combined @ W_fuse + b_fuse; g,b)).  32 rows / WG.
// A (already bf16) staged via GLOBAL_LOAD_ASYNC_TO_LDS_B128 (ASYNCcnt),
// double-buffered with one barrier per k-tile.
// ---------------------------------------------------------------------------
__global__ __launch_bounds__(256) void k_final(
        const unsigned short* __restrict__ combined,
        const unsigned* __restrict__ WfP,
        const float* __restrict__ b_fuse,
        const float* __restrict__ ln_g, const float* __restrict__ ln_b,
        float* __restrict__ item) {
    __shared__ __align__(16) unsigned short sA[2][32 * 40];
    __shared__ float sOut[32 * 256];
    __shared__ float sRed[32 * 8], sRed2[32 * 8];

    int tid = threadIdx.x, lane = tid & 31, w = tid >> 5;
    int rg = w >> 2, cg = w & 3;
    int row0 = blockIdx.x * 32;

    v8f acc[4];
#pragma unroll
    for (int j = 0; j < 4; ++j) acc[j] = zero8();

    async_stage_bf16(combined, FUSE_K, row0, 0, sA[0], 32, tid, 256);
    wait_async0();
    __syncthreads();

    const int KT = FUSE_K / 32;
    for (int kt = 0; kt < KT; ++kt) {
        const unsigned short* cur = sA[kt & 1];
        if (kt + 1 < KT)
            async_stage_bf16(combined, FUSE_K, row0, (kt + 1) * 32, sA[(kt + 1) & 1], 32, tid, 256);
        Frag a; load_a_frag(cur, rg * 16, lane, a);
        Frag b[4];
#pragma unroll
        for (int j = 0; j < 4; ++j)
            load_b_frag(WfP, kt * 16 + (cg * 4 + j), lane, b[j]);
#pragma unroll
        for (int j = 0; j < 4; ++j)
            acc[j] = WMMA_BF16(a.v, b[j].v, acc[j]);
        wait_async0();
        __syncthreads();
    }
    int rbase = rg * 16 + ((lane >> 4) ? 8 : 0);
#pragma unroll
    for (int j = 0; j < 4; ++j) {
        int n = (cg * 4 + j) * 16 + (lane & 15);
#pragma unroll
        for (int v = 0; v < 8; ++v)
            sOut[(rbase + v) * 256 + n] = acc[j][v] + b_fuse[n];
    }
    __syncthreads();

    int r = tid >> 3, seg = tid & 7;
    float s = 0.f, s2 = 0.f;
    for (int c = seg * 32; c < seg * 32 + 32; ++c) {
        float x = sOut[r * 256 + c];
        s += x; s2 += x * x;
    }
    sRed[r * 8 + seg] = s; sRed2[r * 8 + seg] = s2;
    __syncthreads();
    float ts = 0.f, ts2 = 0.f;
#pragma unroll
    for (int i = 0; i < 8; ++i) { ts += sRed[r * 8 + i]; ts2 += sRed2[r * 8 + i]; }
    float mu  = ts * (1.0f / 256.0f);
    float var = ts2 * (1.0f / 256.0f) - mu * mu;
    float inv = rsqrtf(var + 1e-5f);
    size_t gb = (size_t)(row0 + r) * 256;
    for (int c = seg * 32; c < seg * 32 + 32; ++c) {
        float x = (sOut[r * 256 + c] - mu) * inv * ln_g[c] + ln_b[c];
        float g = 0.5f * x * (1.0f + tanhf(0.7978845608028654f * (x + 0.044715f * x * x * x)));
        item[gb + c] = g;
    }
}

// ---------------------------------------------------------------------------
// Kernel: finalize scalar losses (balance entropy from histograms).
// ---------------------------------------------------------------------------
__global__ __launch_bounds__(1024) void k_loss(const int* __restrict__ hist,
                                               const float* __restrict__ accum,
                                               float* __restrict__ out_scalars) {
    __shared__ float red[1024];
    int tid = threadIdx.x;
    float bal = 0.f;
    const float max_ent = logf((float)CB);
    for (int l = 0; l < L_ID; ++l) {
        float cnt  = (float)hist[l * CB + tid];
        float freq = cnt / ((float)NROWS + 1e-8f);
        red[tid] = -freq * logf(freq + 1e-8f);
        __syncthreads();
        for (int s = 512; s > 0; s >>= 1) {
            if (tid < s) red[tid] += red[tid + s];
            __syncthreads();
        }
        if (tid == 0) bal += 1.0f - red[0] / max_ent;
        __syncthreads();
    }
    if (tid == 0) {
        float denom = (float)NROWS * (float)HID;
        out_scalars[0] = accum[1] / denom;                          // recon
        out_scalars[1] = 1.25f * accum[0] / ((float)L_ID * denom);  // residual
        out_scalars[2] = bal / (float)L_ID;                         // balance
    }
}

// ---------------------------------------------------------------------------
// Host-side launch
// ---------------------------------------------------------------------------
extern "C" void kernel_launch(void* const* d_in, const int* in_sizes, int n_in,
                              void* d_out, int out_size, void* d_ws, size_t ws_size,
                              hipStream_t stream) {
    (void)in_sizes; (void)n_in; (void)out_size; (void)ws_size;
    const float* text      = (const float*)d_in[0];
    const float* vis       = (const float*)d_in[1];
    const float* W_text    = (const float*)d_in[2];
    const float* b_text    = (const float*)d_in[3];
    const float* W_vis     = (const float*)d_in[4];
    const float* b_vis     = (const float*)d_in[5];
    const float* modal     = (const float*)d_in[6];
    const float* codebooks = (const float*)d_in[7];
    const float* W_fuse    = (const float*)d_in[8];
    const float* b_fuse    = (const float*)d_in[9];
    const float* ln_g      = (const float*)d_in[10];
    const float* ln_b      = (const float*)d_in[11];

    const size_t NH = (size_t)NROWS * HID;

    // d_out layout: item[NH] | quantized[NH] | recon,res,bal | ids[N*4] (int)
    float* out       = (float*)d_out;
    float* item      = out;
    float* quantized = out + NH;          // used as quant accumulator, then y
    float* scalars   = out + 2 * NH;
    int*   ids       = (int*)(out + 2 * NH + 3);
    float* residual  = item;              // item region doubles as residual scratch

    // workspace layout
    float*          ws       = (float*)d_ws;
    float*          fused    = ws;                                   // NH f32
    unsigned short* combined = (unsigned short*)(ws + NH);           // N*512 bf16
    unsigned*       packW    = (unsigned*)(ws + 2 * NH);
    unsigned* WtP = packW;                       // 24*16*256 = 98304 dwords
    unsigned* WvP = WtP + 98304;                 // 16*16*256 = 65536
    unsigned* WfP = WvP + 65536;                 // 65536
    unsigned* cbP = WfP + 65536;                 // 4 * 8*64*256 = 524288
    float* cnorm  = (float*)(cbP + 524288);      // 4096
    float* accum  = cnorm + 4096;                // [0]=res_sum [1]=recon_sum
    int*   hist   = (int*)(accum + 16);          // 4096

    // 0. zero accumulators / histograms
    k_zero<<<16, 256, 0, stream>>>(hist, accum);

    // 1. pack weights / codebooks into WMMA B-fragment order (bf16)
    k_pack<<<(TXT_K / 32) * (HID / 16), 256, 0, stream>>>(W_text, HID, 1, HID / 16, WtP);
    k_pack<<<(VIS_K / 32) * (HID / 16), 256, 0, stream>>>(W_vis,  HID, 1, HID / 16, WvP);
    k_pack<<<(FUSE_K / 32) * (HID / 16), 256, 0, stream>>>(W_fuse, HID, 1, HID / 16, WfP);
    for (int l = 0; l < L_ID; ++l) {
        // B(k,n) = C[n][k]  -> ldK = 1, ldN = HID
        k_pack<<<(HID / 32) * (CB / 16), 256, 0, stream>>>(
            codebooks + (size_t)l * CB * HID, 1, HID, CB / 16,
            cbP + (size_t)l * (HID / 32) * (CB / 16) * 256);
    }
    k_cnorm<<<L_ID * CB / 256, 256, 0, stream>>>(codebooks, cnorm);

    // 2. encoder + fusion + L2 normalize
    k_encoder<<<NROWS / 32, 256, 0, stream>>>(text, vis, WtP, WvP, b_text, b_vis,
                                              modal, fused, residual, quantized);

    // 3. residual VQ, 4 layers
    for (int l = 0; l < L_ID; ++l) {
        k_rvq<<<NROWS / 32, 256, 0, stream>>>(
            cbP + (size_t)l * (HID / 32) * (CB / 16) * 256,
            codebooks + (size_t)l * CB * HID,
            cnorm + (size_t)l * CB,
            residual, quantized, ids, l, accum + 0, hist + l * CB);
    }

    // 4. mix + LN (no affine) + recon loss + combined bf16
    k_mix<<<NROWS / 32, 256, 0, stream>>>(fused, quantized, combined, accum + 1);

    // 5. final projection + affine LN + gelu
    k_final<<<NROWS / 32, 256, 0, stream>>>(combined, WfP, b_fuse, ln_g, ln_b, item);

    // 6. scalar losses
    k_loss<<<1, 1024, 0, stream>>>(hist, accum, scalars);
}
